// SSTGFAttention_2473901162870
// MI455X (gfx1250) — matrix operand order repaired and checked by reading, not compiled, not run on hardware
//
#include <hip/hip_runtime.h>
#include <hip/hip_bf16.h>

typedef __attribute__((ext_vector_type(16))) _Float16 v16h;
typedef __attribute__((ext_vector_type(8)))  float    v8f;

// Problem constants (match reference setup_inputs)
static constexpr int B_  = 2;
static constexpr int D_  = 8;
static constexpr int H_  = 64;
static constexpr int W_  = 64;
static constexpr int C_  = 96;
static constexpr int NH_ = 4;
static constexpr int HD_ = 24;          // head dim
static constexpr int W2_ = W_ / 2;      // windows along W (split=2)
static constexpr int NWIN_ = B_ * D_ * W2_;   // 512 windows
static constexpr int N_  = 128;         // tokens per window (1*H*2)
static constexpr float SCALE_ = 0.1020620726159658f;  // (384/4)^-0.5 = 1/sqrt(96)

// ---------------- WMMA fragment loaders (contiguous k = 16*(lane/16) + e in both A and B, wave32) ----------------

// A matrix 16x32 (MxK), f16, row-major storage buf[m*rs + k].
__device__ __forceinline__ v16h load_a_frag(const _Float16* buf, int rs, int m0, int k0, int lane) {
  const int m  = m0 + (lane & 15);
  const int kh = (lane >> 4) << 4;       // 0 or 16
  v16h a;
#pragma unroll
  for (int e = 0; e < 16; ++e) {
    const int kk = k0 + kh + e;
    a[e] = buf[m * rs + kk];
  }
  return a;
}

// B matrix 32x16 (KxN), f16, element (k,n) at buf[n*rs + k]  (N-major storage: K^T, W^T).
__device__ __forceinline__ v16h load_bt_frag(const _Float16* buf, int rs, int n0, int k0, int lane) {
  const int n  = n0 + (lane & 15);
  const int kb = k0 + ((lane >> 4) << 4);  // 0 or 16
  v16h b;
#pragma unroll
  for (int e = 0; e < 16; ++e) b[e] = buf[n * rs + kb + e];
  return b;
}

// B matrix 32x16 (KxN), f16, natural row-major storage buf[k*rs + n] (used for V).
__device__ __forceinline__ v16h load_b_frag(const _Float16* buf, int rs, int n0, int k0, int lane) {
  const int n  = n0 + (lane & 15);
  const int kb = k0 + ((lane >> 4) << 4);
  v16h b;
#pragma unroll
  for (int e = 0; e < 16; ++e) b[e] = buf[(kb + e) * rs + n];
  return b;
}

// Convert float4 -> 4 f16 in LDS (merges to ds_store_b64).
__device__ __forceinline__ void st4h(_Float16* dst, float4 f) {
  dst[0] = (_Float16)f.x; dst[1] = (_Float16)f.y;
  dst[2] = (_Float16)f.z; dst[3] = (_Float16)f.w;
}

// ---------------- Kernel 1: windowed attention + LePE, one WG per (window, head) ----------------

__global__ __launch_bounds__(256)
void attn_lepe_kernel(const float* __restrict__ qkv,
                      const float* __restrict__ conv_w,
                      const float* __restrict__ conv_b,
                      float* __restrict__ X /* (NWIN*128, 96) pre-projection tokens */) {
  __shared__ _Float16 Qlds[N_ * 32];   // K-dim padded 24 -> 32 (zeros)
  __shared__ _Float16 Klds[N_ * 32];
  __shared__ _Float16 Vlds[N_ * 32];
  __shared__ _Float16 Plds[N_ * N_];   // softmax probabilities, f16
  __shared__ _Float16 LpS[N_ * HD_];   // LePE conv results for this (window, head)

  const int wg   = blockIdx.x;
  const int win  = wg >> 2;
  const int head = wg & 3;
  const int tid  = threadIdx.x;
  const int lane = tid & 31;
  const int wave = tid >> 5;

  const int b    = win >> 8;           // D_*W2_ = 256
  const int rem  = win & 255;
  const int dpos = rem >> 5;           // W2_ = 32
  const int wblk = rem & 31;

  const unsigned tot = (unsigned)B_ * D_ * H_ * W_ * C_;
  const float* q = qkv;
  const float* k = qkv + tot;
  const float* v = qkv + 2u * tot;

  // ---- stage Q (pre-scaled), K, V as f16 (float4 -> b64 LDS stores), zero pad cols 24..31 ----
#pragma unroll 1
  for (int idx = tid; idx < N_ * 8; idx += 256) {       // 8 quads of 4 cols per token
    const int tok = idx >> 3, qd = idx & 7;
    const int hrow = tok >> 1, wcol = wblk * 2 + (tok & 1);
    _Float16* dq = Qlds + tok * 32 + qd * 4;
    _Float16* dk = Klds + tok * 32 + qd * 4;
    _Float16* dv = Vlds + tok * 32 + qd * 4;
    if (qd < 6) {
      const unsigned g = ((((unsigned)b * D_ + dpos) * H_ + hrow) * W_ + wcol) * C_
                         + head * HD_ + qd * 4;          // 16B aligned
      float4 fq = *reinterpret_cast<const float4*>(q + g);
      float4 fk = *reinterpret_cast<const float4*>(k + g);
      float4 fv = *reinterpret_cast<const float4*>(v + g);
      fq.x *= SCALE_; fq.y *= SCALE_; fq.z *= SCALE_; fq.w *= SCALE_;
      st4h(dq, fq); st4h(dk, fk); st4h(dv, fv);
    } else {
      const float4 z = {0.f, 0.f, 0.f, 0.f};
      st4h(dq, z); st4h(dk, z); st4h(dv, z);
    }
  }

  // ---- LePE pre-pass: depthwise 3x3x3 conv of V, one rolled instance (12 elems/thread) ----
#pragma unroll 1
  for (int idx = tid; idx < N_ * HD_; idx += 256) {
    const int tok = idx / HD_;
    const int dd  = idx - tok * HD_;
    const int ch  = head * HD_ + dd;
    const int hrow = tok >> 1, wcol = wblk * 2 + (tok & 1);
    const float* wc = conv_w + ch * 27;
    float lep = conv_b[ch];
    for (int kd = 0; kd < 3; ++kd) {
      const int dz = dpos + kd - 1; if (dz < 0 || dz >= D_) continue;
      for (int kh = 0; kh < 3; ++kh) {
        const int hz = hrow + kh - 1; if (hz < 0 || hz >= H_) continue;
        for (int kw = 0; kw < 3; ++kw) {
          const int wz = wcol + kw - 1; if (wz < 0 || wz >= W_) continue;
          lep += v[((((unsigned)b * D_ + dz) * H_ + hz) * W_ + wz) * C_ + ch] * wc[kd * 9 + kh * 3 + kw];
        }
      }
    }
    LpS[idx] = (_Float16)lep;
  }
  __syncthreads();

  // ---- GEMM1: S = Q K^T  (each wave owns rows [wave*16, wave*16+16)) ----
  const v16h aQ = load_a_frag(Qlds, 32, wave * 16, 0, lane);
  v8f acc[8];
#pragma unroll
  for (int t = 0; t < 8; ++t) {
    const v16h bK = load_bt_frag(Klds, 32, t * 16, 0, lane);
    v8f z = {};
    acc[t] = __builtin_amdgcn_wmma_f32_16x16x32_f16(false, aQ, false, bK, (short)0, z, false, false);
  }

  // ---- softmax in accumulator registers (row m: VGPR m%8, lane-half m/8) ----
  const int half8 = (lane >> 4) << 3;
  const int coln  = lane & 15;
#pragma unroll
  for (int r = 0; r < 8; ++r) {
    float mx = -3.402823466e38f;
#pragma unroll
    for (int t = 0; t < 8; ++t) mx = fmaxf(mx, acc[t][r]);
#pragma unroll
    for (int off = 1; off < 16; off <<= 1) mx = fmaxf(mx, __shfl_xor(mx, off, 32));
    float s = 0.f;
#pragma unroll
    for (int t = 0; t < 8; ++t) { const float e = __expf(acc[t][r] - mx); acc[t][r] = e; s += e; }
#pragma unroll
    for (int off = 1; off < 16; off <<= 1) s += __shfl_xor(s, off, 32);
    const float inv = 1.f / s;
    const int m = wave * 16 + r + half8;
#pragma unroll
    for (int t = 0; t < 8; ++t)
      Plds[m * N_ + t * 16 + coln] = (_Float16)(acc[t][r] * inv);
  }
  // no barrier: each wave only consumes the P rows it produced

  // ---- GEMM2: O = P V  (head dim 24 -> two 16-wide N tiles, cols 24..31 are zero) ----
  v8f oacc[2];
  oacc[0] = {}; oacc[1] = {};
#pragma unroll
  for (int kc = 0; kc < 4; ++kc) {
    const v16h aP = load_a_frag(Plds, N_, wave * 16, kc * 32, lane);
#pragma unroll
    for (int t = 0; t < 2; ++t) {
      const v16h bV = load_b_frag(Vlds, 32, t * 16, kc * 32, lane);
      oacc[t] = __builtin_amdgcn_wmma_f32_16x16x32_f16(false, aP, false, bV, (short)0, oacc[t], false, false);
    }
  }

  // ---- write-back: O + LePE ----
#pragma unroll
  for (int t = 0; t < 2; ++t) {
    const int dd = t * 16 + coln;
    if (dd >= HD_) continue;
#pragma unroll
    for (int r = 0; r < 8; ++r) {
      const int m = wave * 16 + r + half8;
      X[((unsigned)win * N_ + m) * C_ + head * HD_ + dd] =
          oacc[t][r] + (float)LpS[m * HD_ + dd];
    }
  }
}

// ---------------- Kernel 2: output projection (96x96) + scatter to (B,D,H,W,C) ----------------

__global__ __launch_bounds__(256)
void proj_kernel(const float* __restrict__ X,
                 const float* __restrict__ proj_w,   // (out=96, in=96) row-major
                 const float* __restrict__ proj_b,
                 float* __restrict__ out) {
  __shared__ _Float16 Xl[128 * 96];
  __shared__ _Float16 Wl[96 * 96];

  const int tid = threadIdx.x, lane = tid & 31, wave = tid >> 5;
  const int wg  = blockIdx.x;

#pragma unroll 1
  for (int idx = tid; idx < 128 * 96 / 4; idx += 256) {
    const float4 f = *reinterpret_cast<const float4*>(X + (unsigned)wg * 128 * 96 + idx * 4);
    st4h(Xl + idx * 4, f);
  }
#pragma unroll 1
  for (int idx = tid; idx < 96 * 96 / 4; idx += 256) {
    const float4 f = *reinterpret_cast<const float4*>(proj_w + idx * 4);
    st4h(Wl + idx * 4, f);
  }
  __syncthreads();

  v8f acc[6];
#pragma unroll
  for (int t = 0; t < 6; ++t) acc[t] = {};

#pragma unroll
  for (int kc = 0; kc < 3; ++kc) {                       // K = 96 = 3 x 32
    const v16h aX = load_a_frag(Xl, 96, wave * 16, kc * 32, lane);
#pragma unroll
    for (int t = 0; t < 6; ++t) {                        // N = 96 = 6 x 16
      const v16h bW = load_bt_frag(Wl, 96, t * 16, kc * 32, lane);  // Y = X * W^T
      acc[t] = __builtin_amdgcn_wmma_f32_16x16x32_f16(false, aX, false, bW, (short)0, acc[t], false, false);
    }
  }

  const int half8 = (lane >> 4) << 3;
  const int coln  = lane & 15;
#pragma unroll
  for (int t = 0; t < 6; ++t) {
    const int o = t * 16 + coln;
    const float bias = proj_b[o];
#pragma unroll
    for (int r = 0; r < 8; ++r) {
      const int m = wave * 16 + r + half8;
      const unsigned tok = (unsigned)wg * 128 + m;
      const int win  = (int)(tok >> 7);
      const int n    = (int)(tok & 127);
      const int bb   = win >> 8;         // / (D_*W2_) = /256
      const int rem  = win & 255;
      const int dpos = rem >> 5;         // / W2_ = /32
      const int wblk = rem & 31;
      const int hrow = n >> 1;
      const int wcol = wblk * 2 + (n & 1);
      out[((((unsigned)bb * D_ + dpos) * H_ + hrow) * W_ + wcol) * C_ + o] = acc[t][r] + bias;
    }
  }
}

// ---------------- host-side launcher ----------------

extern "C" void kernel_launch(void* const* d_in, const int* in_sizes, int n_in,
                              void* d_out, int out_size, void* d_ws, size_t ws_size,
                              hipStream_t stream) {
  (void)in_sizes; (void)n_in; (void)out_size; (void)ws_size;
  const float* qkv    = (const float*)d_in[0];
  const float* conv_w = (const float*)d_in[1];
  const float* conv_b = (const float*)d_in[2];
  const float* proj_w = (const float*)d_in[3];
  const float* proj_b = (const float*)d_in[4];
  float* out = (float*)d_out;
  float* X   = (float*)d_ws;   // (512*128, 96) f32 pre-projection tokens = 24 MB

  attn_lepe_kernel<<<NWIN_ * NH_, 256, 0, stream>>>(qkv, conv_w, conv_b, X);
  proj_kernel<<<NWIN_ * N_ / 128, 256, 0, stream>>>(X, proj_w, proj_b, out);
}
